// PNALayer_27736898797955
// MI455X (gfx1250) — matrix operand order repaired and checked by reading, not compiled, not run on hardware
//
#include <hip/hip_runtime.h>
#include <cstdint>

// ---------------- problem constants ----------------
constexpr int Bc  = 4, Cc = 128, Hc = 48, Wc = 48, NHc = 4, HDc = 32, Kc = 7;
constexpr int HWc = Hc * Wc;          // 2304
constexpr int Sc  = HWc;              // 2304 tokens
constexpr int Mc  = Bc * Sc;          // 9216 rows
constexpr float SCALEc = 0.17677669529663687f;   // HD^-0.5

// ---------------- WMMA fragment types ----------------
typedef _Float16 h16v __attribute__((ext_vector_type(16)));
typedef _Float16 h8v  __attribute__((ext_vector_type(8)));
typedef float    f8v  __attribute__((ext_vector_type(8)));

union H16U { h16v v; h8v h[2]; };

static __device__ __forceinline__ h16v load_frag(const _Float16* p0, const _Float16* p1) {
  H16U u;
  u.h[0] = *reinterpret_cast<const h8v*>(p0);
  u.h[1] = *reinterpret_cast<const h8v*>(p1);
  return u.v;
}

static __device__ __forceinline__ float wred_sum(float v) {
#pragma unroll
  for (int m = 16; m >= 1; m >>= 1) v += __shfl_xor(v, m, 32);
  return v;
}

static __device__ __forceinline__ void async_b128(unsigned ldsOff, const void* g) {
  // gfx1250 async global->LDS copy, tracked by ASYNCcnt
  asm volatile("global_load_async_to_lds_b128 %0, %1, off"
               :: "v"(ldsOff), "v"((unsigned long long)(uintptr_t)g) : "memory");
}

// ---------------- weight convert: f32 (K x N) -> f16 transposed (N x K) ----------------
__global__ void conv_wT(const float* __restrict__ w, _Float16* __restrict__ wt, int Kk, int Nn) {
  int idx = blockIdx.x * blockDim.x + threadIdx.x;
  if (idx >= Kk * Nn) return;
  int n = idx % Nn, k = idx / Nn;
  wt[(size_t)n * Kk + k] = (_Float16)w[idx];
}

// ---------------- generic WMMA GEMM: C = epilogue(A[f16 MxK] * Wt[f16 NxK]^T) ----------------
__global__ __launch_bounds__(256)
void gemm16_wmma(const _Float16* __restrict__ A, const _Float16* __restrict__ Wt,
                 const float* __restrict__ bias, const float* __restrict__ res,
                 float* __restrict__ outF, _Float16* __restrict__ outH,
                 int Mm, int Nn, int Kk, float scale, int gelu) {
  const int lane = threadIdx.x & 31;
  const int wave = (blockIdx.x << 3) + (threadIdx.x >> 5);
  const int ntiles = Nn >> 4;
  const int m0 = (wave / ntiles) << 4;
  const int n0 = (wave % ntiles) << 4;
  if (m0 >= Mm) return;
  const int row = lane & 15, hi = lane >> 4;

  // A fragment: lane row = lane&15; halves 0-7 -> K=base..base+7, halves 8-15 -> K=base+16..base+23, base=hi*8
  const _Float16* Ap = A  + (size_t)(m0 + row) * Kk + hi * 8;
  // B fragment: lane col N = n0+(lane&15); halves 0-7 -> K=base..base+7, halves 8-15 -> K=base+16..base+23, base=hi*8
  const _Float16* Wp = Wt + (size_t)(n0 + row) * Kk + hi * 8;

  f8v acc = {};
#pragma unroll 4
  for (int k0 = 0; k0 < Kk; k0 += 32) {
    h16v a = load_frag(Ap + k0, Ap + k0 + 16);
    h16v b = load_frag(Wp + k0, Wp + k0 + 16);
    acc = __builtin_amdgcn_wmma_f32_16x16x32_f16(false, a, false, b, (short)0, acc, false, false);
  }
  const int n = n0 + row;
  const float bn = bias ? bias[n] : 0.0f;
#pragma unroll
  for (int r = 0; r < 8; ++r) {
    const int m = m0 + r + hi * 8;                 // C layout: vgpr r, lanes 0-15 -> M=r, 16-31 -> M=8+r
    float xv = (acc[r] + bn) * scale;
    if (gelu) xv = 0.5f * xv * (1.0f + erff(xv * 0.70710678118f));
    if (res)  xv += res[(size_t)m * Nn + n];
    if (outF) outF[(size_t)m * Nn + n] = xv;
    if (outH) outH[(size_t)m * Nn + n] = (_Float16)xv;
  }
}

// ---------------- LayerNorm over C=128, input BCHW (fused transpose) ----------------
__global__ __launch_bounds__(256)
void ln_bchw(const float* __restrict__ src, const float* __restrict__ g, const float* __restrict__ bb,
             float* __restrict__ rawF, _Float16* __restrict__ outH) {
  const int row  = blockIdx.x * 8 + (threadIdx.x >> 5);
  const int lane = threadIdx.x & 31;
  if (row >= Mc) return;
  const int b = row / Sc, s = row % Sc;
  const float* base = src + (size_t)b * Cc * HWc + s;
  float xv[4], sum = 0.f, sq = 0.f;
#pragma unroll
  for (int i = 0; i < 4; ++i) {
    int c = lane + 32 * i;
    xv[i] = base[(size_t)c * HWc];
    sum += xv[i]; sq += xv[i] * xv[i];
  }
  sum = wred_sum(sum); sq = wred_sum(sq);
  float mean = sum * (1.0f / Cc);
  float inv  = rsqrtf(sq * (1.0f / Cc) - mean * mean + 1e-5f);
#pragma unroll
  for (int i = 0; i < 4; ++i) {
    int c = lane + 32 * i;
    float y = (xv[i] - mean) * inv * g[c] + bb[c];
    outH[(size_t)row * Cc + c] = (_Float16)y;
    if (rawF) rawF[(size_t)row * Cc + c] = xv[i];
  }
}

// ---------------- LayerNorm over C=128, input row-major (M,C) ----------------
__global__ __launch_bounds__(256)
void ln_rowmajor(const float* __restrict__ src, const float* __restrict__ g, const float* __restrict__ bb,
                 _Float16* __restrict__ outH, float* __restrict__ outBCHW) {
  const int row  = blockIdx.x * 8 + (threadIdx.x >> 5);
  const int lane = threadIdx.x & 31;
  if (row >= Mc) return;
  const int b = row / Sc, s = row % Sc;
  float xv[4], sum = 0.f, sq = 0.f;
#pragma unroll
  for (int i = 0; i < 4; ++i) {
    int c = lane + 32 * i;
    xv[i] = src[(size_t)row * Cc + c];
    sum += xv[i]; sq += xv[i] * xv[i];
  }
  sum = wred_sum(sum); sq = wred_sum(sq);
  float mean = sum * (1.0f / Cc);
  float inv  = rsqrtf(sq * (1.0f / Cc) - mean * mean + 1e-5f);
#pragma unroll
  for (int i = 0; i < 4; ++i) {
    int c = lane + 32 * i;
    float y = (xv[i] - mean) * inv * g[c] + bb[c];
    if (outH)    outH[(size_t)row * Cc + c] = (_Float16)y;
    if (outBCHW) outBCHW[(size_t)(b * Cc + c) * HWc + s] = y;
  }
}

// ---------------- repack v (cols 256..383 of qkv) to vT[b,nh,hd,s] ----------------
__global__ void repack_vT(const _Float16* __restrict__ qkv, _Float16* __restrict__ vT) {
  int idx = blockIdx.x * blockDim.x + threadIdx.x;
  if (idx >= Mc * Cc) return;
  int d = idx & 31, nh = (idx >> 5) & 3, sr = idx >> 7;
  int s = sr % Sc, b = sr / Sc;
  vT[((size_t)((b * NHc + nh) * HDc + d)) * Sc + s] = qkv[(size_t)sr * 384 + 256 + nh * 32 + d];
}

// ---------------- flash self-attention: 1 wave per 16-query strip, 4 strips/block ----------------
// LDS (halves): k[2][32x32] @0, vT-chunk[2][32x32] @2048, P[4 waves][16x32] @4096  => 12288 bytes
__global__ __launch_bounds__(128)
void flash_attn(const _Float16* __restrict__ qkv, const _Float16* __restrict__ vT,
                _Float16* __restrict__ oatt) {
  extern __shared__ _Float16 lds[];
  const int tid = threadIdx.x;
  const int wave = tid >> 5, lane = tid & 31;
  const int row = lane & 15, hi = lane >> 4;
  int grp = blockIdx.x;
  const int sg = grp % 36; grp /= 36;
  const int nh = grp % NHc; const int b = grp / NHc;
  const int m0 = (sg * 4 + wave) * 16;

  // q A-fragment (16x32, HD=32), stays in registers for whole strip
  const _Float16* qrow = qkv + (size_t)(b * Sc + m0 + row) * 384 + nh * 32 + hi * 8;
  h16v qf = load_frag(qrow, qrow + 16);

  const int tok = tid >> 2, seg = tid & 3;   // 128 threads x 16B = one 2KB chunk each for k and v
  auto issue = [&](int j, int db) {
    int t0 = j * 32;
    const _Float16* ks = qkv + (size_t)(b * Sc + t0 + tok) * 384 + 128 + nh * 32 + seg * 8;
    async_b128((unsigned)((db * 1024 + tok * 32 + seg * 8) * 2), ks);
    const _Float16* vs = vT + (size_t)((b * NHc + nh) * HDc + tok) * Sc + t0 + seg * 8;
    async_b128((unsigned)((2048 + db * 1024 + tok * 32 + seg * 8) * 2), vs);
  };

  float mrow[8], lrow[8];
#pragma unroll
  for (int r = 0; r < 8; ++r) { mrow[r] = -3.0e38f; lrow[r] = 0.0f; }
  f8v o0 = {}, o1 = {};
  _Float16* pbuf = lds + 4096 + wave * 512;

  constexpr int NC = Sc / 32;   // 72 chunks of 32 tokens
  issue(0, 0);
  for (int j = 0; j < NC; ++j) {
    const int db = j & 1;
    if (j + 1 < NC) { issue(j + 1, db ^ 1); asm volatile("s_wait_asynccnt 2" ::: "memory"); }
    else            {                        asm volatile("s_wait_asynccnt 0" ::: "memory"); }
    __syncthreads();

    const _Float16* kc = lds + db * 1024;          // token-major [32 tok][32 dim]
    const _Float16* vc = lds + 2048 + db * 1024;   // dim-major   [32 dim][32 tok]

    // scores: B-fragment lane: token = tile+row, K = hi*8+h (h<8), 16+hi*8+(h-8) (h>=8)
    const _Float16* kb0 = kc + row * 32 + hi * 8;
    const _Float16* kb1 = kc + (16 + row) * 32 + hi * 8;
    f8v z = {};
    f8v s0 = __builtin_amdgcn_wmma_f32_16x16x32_f16(false, qf, false, load_frag(kb0, kb0 + 16), (short)0, z, false, false);
    f8v s1 = __builtin_amdgcn_wmma_f32_16x16x32_f16(false, qf, false, load_frag(kb1, kb1 + 16), (short)0, z, false, false);

#pragma unroll
    for (int r = 0; r < 8; ++r) {
      float a0 = s0[r] * SCALEc, a1 = s1[r] * SCALEc;
      float tm = fmaxf(a0, a1);
#pragma unroll
      for (int m = 8; m >= 1; m >>= 1) tm = fmaxf(tm, __shfl_xor(tm, m, 32));
      float nm = fmaxf(mrow[r], tm);
      float alpha = __expf(mrow[r] - nm);
      float p0 = __expf(a0 - nm), p1 = __expf(a1 - nm);
      float ps = p0 + p1;
#pragma unroll
      for (int m = 8; m >= 1; m >>= 1) ps += __shfl_xor(ps, m, 32);
      lrow[r] = lrow[r] * alpha + ps;
      o0[r] *= alpha; o1[r] *= alpha;
      mrow[r] = nm;
      pbuf[(r + hi * 8) * 32 + row]      = (_Float16)p0;   // P row-major 16x32 (C->A relayout via LDS)
      pbuf[(r + hi * 8) * 32 + 16 + row] = (_Float16)p1;
    }
    asm volatile("s_wait_dscnt 0" ::: "memory");
    __builtin_amdgcn_wave_barrier();

    const _Float16* pr = pbuf + row * 32 + hi * 8;
    h16v pf = load_frag(pr, pr + 16);
    const _Float16* vb0 = vc + row * 32 + hi * 8;          // N=dim, K=token split hi*8 / 16+hi*8
    const _Float16* vb1 = vc + (16 + row) * 32 + hi * 8;
    o0 = __builtin_amdgcn_wmma_f32_16x16x32_f16(false, pf, false, load_frag(vb0, vb0 + 16), (short)0, o0, false, false);
    o1 = __builtin_amdgcn_wmma_f32_16x16x32_f16(false, pf, false, load_frag(vb1, vb1 + 16), (short)0, o1, false, false);
    __syncthreads();   // all waves done with chunk j before its buffer is re-filled
  }

#pragma unroll
  for (int r = 0; r < 8; ++r) {
    float inv = 1.0f / lrow[r];
    size_t orow = (size_t)(b * Sc + m0 + r + hi * 8) * Cc + nh * 32;
    oatt[orow + row]      = (_Float16)(o0[r] * inv);
    oatt[orow + 16 + row] = (_Float16)(o1[r] * inv);
  }
}

// ---------------- neighborhood cross-attention (49 neighbors, scalar per (b,h,w,head)) ----------------
__global__ __launch_bounds__(256)
void nbr_attn(const _Float16* __restrict__ qn, const _Float16* __restrict__ kvn,
              const float* __restrict__ rpb, _Float16* __restrict__ on) {
  int idx = blockIdx.x * blockDim.x + threadIdx.x;
  if (idx >= Bc * Hc * Wc * NHc) return;
  const int n = idx & 3;
  int t = idx >> 2;
  const int w = t % Wc; t /= Wc;
  const int h = t % Hc;
  const int b = t / Hc;
  const int sh = min(max(h - 3, 0), Hc - Kc);
  const int sw = min(max(w - 3, 0), Wc - Kc);
  const int row = (b * Hc + h) * Wc + w;

  float q[HDc];
#pragma unroll
  for (int d = 0; d < HDc; ++d) q[d] = (float)qn[(size_t)row * Cc + n * HDc + d];  // SCALE pre-folded

  float mx = -3.0e38f;
  for (int i = 0; i < Kc; ++i)
    for (int j = 0; j < Kc; ++j) {
      int nr = (b * Hc + sh + i) * Wc + (sw + j);
      const _Float16* kp = kvn + (size_t)nr * 256 + n * HDc;
      float dt = 0.f;
#pragma unroll
      for (int d = 0; d < HDc; ++d) dt += q[d] * (float)kp[d];
      dt += rpb[(n * 13 + (6 + i - (h - sh))) * 13 + (6 + j - (w - sw))];
      mx = fmaxf(mx, dt);
    }

  float sum = 0.f, acc[HDc];
#pragma unroll
  for (int d = 0; d < HDc; ++d) acc[d] = 0.f;
  for (int i = 0; i < Kc; ++i)
    for (int j = 0; j < Kc; ++j) {
      int nr = (b * Hc + sh + i) * Wc + (sw + j);
      const _Float16* kp = kvn + (size_t)nr * 256 + n * HDc;
      float dt = 0.f;
#pragma unroll
      for (int d = 0; d < HDc; ++d) dt += q[d] * (float)kp[d];
      dt += rpb[(n * 13 + (6 + i - (h - sh))) * 13 + (6 + j - (w - sw))];
      float p = __expf(dt - mx);
      sum += p;
      const _Float16* vp = kvn + (size_t)nr * 256 + 128 + n * HDc;
#pragma unroll
      for (int d = 0; d < HDc; ++d) acc[d] += p * (float)vp[d];
    }
  float inv = 1.0f / sum;
#pragma unroll
  for (int d = 0; d < HDc; ++d) on[(size_t)row * Cc + n * HDc + d] = (_Float16)(acc[d] * inv);
}

// ---------------- host orchestration ----------------
extern "C" void kernel_launch(void* const* d_in, const int* in_sizes, int n_in,
                              void* d_out, int out_size, void* d_ws, size_t ws_size,
                              hipStream_t stream) {
  (void)in_sizes; (void)n_in; (void)out_size; (void)ws_size;
  const float* x          = (const float*)d_in[0];
  const float* v          = (const float*)d_in[1];
  const float* sa_ln1_g   = (const float*)d_in[2];
  const float* sa_ln1_b   = (const float*)d_in[3];
  const float* sa_qkv_w   = (const float*)d_in[4];
  const float* sa_qkv_b   = (const float*)d_in[5];
  const float* sa_proj_w  = (const float*)d_in[6];
  const float* sa_proj_b  = (const float*)d_in[7];
  const float* sa_ln2_g   = (const float*)d_in[8];
  const float* sa_ln2_b   = (const float*)d_in[9];
  const float* sa_fc1_w   = (const float*)d_in[10];
  const float* sa_fc1_b   = (const float*)d_in[11];
  const float* sa_fc2_w   = (const float*)d_in[12];
  const float* sa_fc2_b   = (const float*)d_in[13];
  const float* n1_g       = (const float*)d_in[14];
  const float* n1_b       = (const float*)d_in[15];
  const float* n12_g      = (const float*)d_in[16];
  const float* n12_b      = (const float*)d_in[17];
  const float* q_w        = (const float*)d_in[18];
  const float* q_b        = (const float*)d_in[19];
  const float* kv_w       = (const float*)d_in[20];
  const float* kv_b       = (const float*)d_in[21];
  const float* rpb        = (const float*)d_in[22];
  const float* att_proj_w = (const float*)d_in[23];
  const float* att_proj_b = (const float*)d_in[24];
  const float* n2_g       = (const float*)d_in[25];
  const float* n2_b       = (const float*)d_in[26];
  const float* fc1_w      = (const float*)d_in[27];
  const float* fc1_b      = (const float*)d_in[28];
  const float* fc2_w      = (const float*)d_in[29];
  const float* fc2_b      = (const float*)d_in[30];
  const float* no_g       = (const float*)d_in[31];
  const float* no_b       = (const float*)d_in[32];
  float* out = (float*)d_out;

  char* ws = (char*)d_ws;
  size_t off = 0;
  auto alloc = [&](size_t bytes) -> char* {
    char* p = ws + off;
    off = (off + bytes + 255) & ~(size_t)255;
    return p;
  };

  // transposed f16 weights
  _Float16* wQKV  = (_Float16*)alloc((size_t)128 * 384 * 2);
  _Float16* wPROJ = (_Float16*)alloc((size_t)128 * 128 * 2);
  _Float16* wFC1S = (_Float16*)alloc((size_t)128 * 512 * 2);
  _Float16* wFC2S = (_Float16*)alloc((size_t)512 * 128 * 2);
  _Float16* wQ    = (_Float16*)alloc((size_t)128 * 128 * 2);
  _Float16* wKV   = (_Float16*)alloc((size_t)128 * 256 * 2);
  _Float16* wAP   = (_Float16*)alloc((size_t)128 * 128 * 2);
  _Float16* wFC1  = (_Float16*)alloc((size_t)128 * 512 * 2);
  _Float16* wFC2  = (_Float16*)alloc((size_t)512 * 128 * 2);
  // f32 activations (M x 128)
  float* T    = (float*)alloc((size_t)Mc * Cc * 4);
  float* T2   = (float*)alloc((size_t)Mc * Cc * 4);
  float* SC   = (float*)alloc((size_t)Mc * Cc * 4);
  float* HSUM = (float*)alloc((size_t)Mc * Cc * 4);
  float* HFIN = (float*)alloc((size_t)Mc * Cc * 4);
  // f16 activations
  _Float16* Hh   = (_Float16*)alloc((size_t)Mc * Cc * 2);
  _Float16* QKV  = (_Float16*)alloc((size_t)Mc * 384 * 2);
  _Float16* VT   = (_Float16*)alloc((size_t)Mc * Cc * 2);
  _Float16* OATT = (_Float16*)alloc((size_t)Mc * Cc * 2);
  _Float16* MLP  = (_Float16*)alloc((size_t)Mc * 512 * 2);
  _Float16* XN   = (_Float16*)alloc((size_t)Mc * Cc * 2);
  _Float16* VN   = (_Float16*)alloc((size_t)Mc * Cc * 2);
  _Float16* QN   = (_Float16*)alloc((size_t)Mc * Cc * 2);
  _Float16* KVN  = (_Float16*)alloc((size_t)Mc * 256 * 2);
  _Float16* ON   = (_Float16*)alloc((size_t)Mc * Cc * 2);

  auto conv = [&](const float* w, _Float16* wt, int Kk, int Nn) {
    int n = Kk * Nn;
    conv_wT<<<(n + 255) / 256, 256, 0, stream>>>(w, wt, Kk, Nn);
  };
  auto gemm = [&](const _Float16* A, const _Float16* Wt, const float* bias, const float* res,
                  float* oF, _Float16* oH, int Mm, int Nn, int Kk, float scale, int gelu) {
    int tiles = (Mm / 16) * (Nn / 16);
    gemm16_wmma<<<tiles / 8, 256, 0, stream>>>(A, Wt, bias, res, oF, oH, Mm, Nn, Kk, scale, gelu);
  };
  const int lnGrid = Mc / 8;

  // weights -> f16 transposed
  conv(sa_qkv_w,  wQKV,  128, 384);
  conv(sa_proj_w, wPROJ, 128, 128);
  conv(sa_fc1_w,  wFC1S, 128, 512);
  conv(sa_fc2_w,  wFC2S, 512, 128);
  conv(q_w,       wQ,    128, 128);
  conv(kv_w,      wKV,   128, 256);
  conv(att_proj_w,wAP,   128, 128);
  conv(fc1_w,     wFC1,  128, 512);
  conv(fc2_w,     wFC2,  512, 128);

  // ---- self-attention block ----
  ln_bchw<<<lnGrid, 256, 0, stream>>>(x, sa_ln1_g, sa_ln1_b, T, Hh);                 // T = x^T, Hh = LN1
  gemm(Hh, wQKV, sa_qkv_b, nullptr, nullptr, QKV, Mc, 384, 128, 1.0f, 0);            // qkv
  repack_vT<<<(Mc * Cc + 255) / 256, 256, 0, stream>>>(QKV, VT);                     // v -> vT
  flash_attn<<<Bc * NHc * 36, 128, 12288, stream>>>(QKV, VT, OATT);                  // attention
  gemm(OATT, wPROJ, sa_proj_b, T, T2, nullptr, Mc, 128, 128, 1.0f, 0);               // T2 = T + proj(o)
  ln_rowmajor<<<lnGrid, 256, 0, stream>>>(T2, sa_ln2_g, sa_ln2_b, Hh, nullptr);      // Hh = LN2
  gemm(Hh, wFC1S, sa_fc1_b, nullptr, nullptr, MLP, Mc, 512, 128, 1.0f, 1);           // gelu(fc1)
  gemm(MLP, wFC2S, sa_fc2_b, T2, SC, nullptr, Mc, 128, 512, 1.0f, 0);                // SC = shortcut

  // ---- neighborhood branch ----
  ln_bchw<<<lnGrid, 256, 0, stream>>>(x, n1_g,  n1_b,  nullptr, XN);                 // xn
  ln_bchw<<<lnGrid, 256, 0, stream>>>(v, n12_g, n12_b, nullptr, VN);                 // vn
  gemm(XN, wQ,  q_b,  nullptr, nullptr, QN,  Mc, 128, 128, SCALEc, 0);               // q (pre-scaled)
  gemm(VN, wKV, kv_b, nullptr, nullptr, KVN, Mc, 256, 128, 1.0f, 0);                 // k|v
  nbr_attn<<<(Bc * Hc * Wc * NHc + 255) / 256, 256, 0, stream>>>(QN, KVN, rpb, ON);  // o_n
  gemm(ON, wAP, att_proj_b, SC, HSUM, nullptr, Mc, 128, 128, 1.0f, 0);               // h = shortcut + a

  // ---- final MLP + output LN ----
  ln_rowmajor<<<lnGrid, 256, 0, stream>>>(HSUM, n2_g, n2_b, Hh, nullptr);            // LN(h)
  gemm(Hh, wFC1, fc1_b, nullptr, nullptr, MLP, Mc, 512, 128, 1.0f, 1);               // gelu(fc1)
  gemm(MLP, wFC2, fc2_b, HSUM, HFIN, nullptr, Mc, 128, 512, 1.0f, 0);                // h += fc2(...)
  ln_rowmajor<<<lnGrid, 256, 0, stream>>>(HFIN, no_g, no_b, nullptr, out);           // out (BCHW)
}
